// YBLoss_84628035600897
// MI455X (gfx1250) — matrix operand order repaired and checked by reading, compiled, not run on hardware
//
#include <hip/hip_runtime.h>
#include <hip/hip_bf16.h>
#include <hip/hip_fp16.h>

typedef __attribute__((ext_vector_type(16))) _Float16 v16h;
typedef __attribute__((ext_vector_type(8)))  float    v8f;

#define TCNT 10
#define KCNT 25
#define ROWF (2 * KCNT * TCNT)   // 500 floats per [T,2,K] row
#define TILE 32                  // rows per block tile
#define THREADS 256

// ---------------------------------------------------------------------------
// Zero-init workspace region (touched flags, neg accumulators, 4 global accs)
// ---------------------------------------------------------------------------
__global__ void k_zero(float* __restrict__ p, int n) {
    for (int i = blockIdx.x * blockDim.x + threadIdx.x; i < n;
         i += gridDim.x * blockDim.x)
        p[i] = 0.0f;
}

// ---------------------------------------------------------------------------
// cond[n] = (dot25(target[aidx[n]], target[vidx[n]]) == 0) via WMMA.
// One wave handles 16 pairs: A = 16 audio rows x 32 (K padded, f16),
// B = 32 x 16 visual rows (f16). D = A*B in f32; diag(D) are the 16 dots.
// 0/1 values are exact in f16; integer sums <= 25 exact in f32.
// ---------------------------------------------------------------------------
__global__ void k_cond(const float* __restrict__ target,
                       const int*   __restrict__ aidx,
                       const int*   __restrict__ vidx,
                       float* __restrict__ cond_f,
                       float* __restrict__ touched_a,
                       float* __restrict__ touched_v,
                       int n_pairs) {
    int wave = (blockIdx.x * blockDim.x + threadIdx.x) >> 5;
    int lane = threadIdx.x & 31;
    int base = wave * 16;                // wave-uniform
    if (base >= n_pairs) return;         // uniform exit: EXEC stays all-1s

    int half = lane >> 4;                // 0: lanes 0-15, 1: lanes 16-31
    int mn   = lane & 15;
    const float* ta = target + (size_t)aidx[base + mn] * KCNT;
    const float* tv = target + (size_t)vidx[base + mn] * KCNT;

    union { v16h v; _Float16 h[16]; } A, Bm;
#pragma unroll
    for (int e = 0; e < 16; ++e) {
        // 16-bit A 16x32 layout: VGPR v<4 -> K = half*8 + 2v(+sub);
        //                        VGPR v>=4 -> K = 16 + half*8 + 2(v-4)(+sub)
        int ka = ((e >= 8) ? 16 : 0) + half * 8 + (e & 7);
        A.h[e] = (ka < KCNT) ? (_Float16)ta[ka] : (_Float16)0.0f;
        // 16-bit B 32x16 layout: same K order as A (e<8: half*8+e; e>=8: 16+half*8+(e-8))
        int kb = ((e >= 8) ? 16 : 0) + half * 8 + (e & 7);
        Bm.h[e] = (kb < KCNT) ? (_Float16)tv[kb] : (_Float16)0.0f;
    }

    v8f c = {};
    c = __builtin_amdgcn_wmma_f32_16x16x32_f16(
            /*neg_a=*/false, A.v, /*neg_b=*/false, Bm.v,
            /*c_mod=*/(short)0, c, /*reuse_a=*/false, /*reuse_b=*/false);

    union { v8f v; float f[8]; } C; C.v = c;
    // diag D[m][m]: m<8 lives at lane m, comp m; m>=8 at lane m+16, comp m-8
    bool valid = (lane < 8) | (lane >= 24);
    if (valid) {
        int   m   = (lane < 8) ? lane : (lane - 16);
        float dot = C.f[lane & 7];
        int   n   = base + m;
        float cf  = (dot == 0.0f) ? 1.0f : 0.0f;
        cond_f[n] = cf;
        if (cf != 0.0f) {                 // idempotent flag stores (race-benign)
            touched_a[aidx[n]] = 1.0f;
            touched_v[vidx[n]] = 1.0f;
        }
    }
}

// ---------------------------------------------------------------------------
// Async global -> LDS tile copy (TILE rows of 2000 B), ASYNCcnt-tracked.
// ---------------------------------------------------------------------------
__device__ __forceinline__ void async_tile_copy(const float* __restrict__ src,
                                                float* __restrict__ tile) {
    unsigned lds0 = (unsigned)(size_t)&tile[0];     // addr[31:0] = LDS offset
    unsigned long long g0 = (unsigned long long)(size_t)src;
    const int chunks = TILE * ROWF * 4 / 16;        // 4000 x 16B
    for (int c = threadIdx.x; c < chunks; c += THREADS) {
        unsigned           lds = lds0 + (unsigned)c * 16u;
        unsigned long long g   = g0 + (unsigned long long)c * 16ull;
        asm volatile("global_load_async_to_lds_b128 %0, %1, off"
                     :: "v"(lds), "v"(g) : "memory");
    }
    asm volatile("s_wait_asynccnt 0" ::: "memory");
    __syncthreads();
}

// ---------------------------------------------------------------------------
// Negative pooling: rows [b .. b+n) of all_prob; exp(max_t), cond-gated
// scatter-add into per-sample neg accumulators.
// ---------------------------------------------------------------------------
__global__ void k_neg(const float* __restrict__ all_prob,
                      const int*   __restrict__ aidx,
                      const int*   __restrict__ vidx,
                      const float* __restrict__ cond_f,
                      float* __restrict__ a_neg,
                      float* __restrict__ v_neg,
                      int b_rows) {
    __shared__ float tile[TILE * ROWF];              // 64000 B
    int row0 = blockIdx.x * TILE;
    async_tile_copy(all_prob + (size_t)(b_rows + row0) * ROWF, tile);

    for (int j = threadIdx.x; j < TILE * 2 * KCNT; j += THREADS) {
        int row = j / (2 * KCNT);
        int rem = j % (2 * KCNT);
        int s   = rem / KCNT;                        // 0 = audio slot, 1 = visual
        int k   = rem % KCNT;
        int n   = row0 + row;
        float cf = cond_f[n];
        if (cf == 0.0f) continue;
        const float* p = &tile[row * ROWF + s * KCNT + k];
        float m = p[0];
#pragma unroll
        for (int t = 1; t < TCNT; ++t) m = fmaxf(m, p[t * 2 * KCNT]);
        float e = expf(m);
        if (s == 1) atomicAdd(&a_neg[(size_t)aidx[n] * KCNT + k], e);
        else        atomicAdd(&v_neg[(size_t)vidx[n] * KCNT + k], e);
    }
}

// ---------------------------------------------------------------------------
// Positive pooling + masked log-ratio accumulation over rows [0, b).
// acc = {sum_a, cnt_a, sum_v, cnt_v} (sums already hold -log(ratio)).
// ---------------------------------------------------------------------------
__global__ void k_pos_loss(const float* __restrict__ all_prob,
                           const float* __restrict__ target,
                           const float* __restrict__ touched_a,
                           const float* __restrict__ touched_v,
                           const float* __restrict__ a_neg,
                           const float* __restrict__ v_neg,
                           float* __restrict__ acc) {
    __shared__ float tile[TILE * ROWF];              // 64000 B
    __shared__ float red[4];
    if (threadIdx.x < 4) red[threadIdx.x] = 0.0f;    // covered by copy barrier
    int row0 = blockIdx.x * TILE;
    async_tile_copy(all_prob + (size_t)row0 * ROWF, tile);

    float loc[4] = {0.0f, 0.0f, 0.0f, 0.0f};         // sum_a,cnt_a,sum_v,cnt_v
    for (int j = threadIdx.x; j < TILE * 2 * KCNT; j += THREADS) {
        int row = j / (2 * KCNT);
        int rem = j % (2 * KCNT);
        int s   = rem / KCNT;                        // 0 -> audio, 1 -> visual
        int k   = rem % KCNT;
        int b   = row0 + row;
        float tg  = target[(size_t)b * KCNT + k];
        float tch = (s == 0) ? touched_a[b] : touched_v[b];
        const float* p = &tile[row * ROWF + s * KCNT + k];
        float m = p[0];
#pragma unroll
        for (int t = 1; t < TCNT; ++t) m = fmaxf(m, p[t * 2 * KCNT]);
        float pos = expf(m) * tch * tg;
        if (pos != 0.0f) {
            float neg  = ((s == 0) ? a_neg : v_neg)[(size_t)b * KCNT + k] * tg;
            float term = logf((pos + neg) / pos);    // == -log(pos/(pos+neg))
            int o = (s == 0) ? 0 : 2;
            loc[o]     += term;
            loc[o + 1] += 1.0f;
        }
    }
    __syncthreads();
#pragma unroll
    for (int i = 0; i < 4; ++i)
        if (loc[i] != 0.0f) atomicAdd(&red[i], loc[i]);   // ds_add_f32
    __syncthreads();
    if (threadIdx.x < 4) atomicAdd(&acc[threadIdx.x], red[threadIdx.x]);
}

__global__ void k_fin(const float* __restrict__ acc, float* __restrict__ out) {
    if (blockIdx.x == 0 && threadIdx.x == 0)
        out[0] = acc[2] / acc[3] + acc[0] / acc[1];  // loss_v + loss_a
}

// ---------------------------------------------------------------------------
extern "C" void kernel_launch(void* const* d_in, const int* in_sizes, int n_in,
                              void* d_out, int out_size, void* d_ws, size_t ws_size,
                              hipStream_t stream) {
    const float* all_prob = (const float*)d_in[0];
    const float* target   = (const float*)d_in[1];
    const int*   aidx     = (const int*)d_in[2];
    const int*   vidx     = (const int*)d_in[3];

    const int n = in_sizes[2];            // N pairs
    const int b = in_sizes[1] / KCNT;     // B samples

    // workspace layout (floats): cond_f[n] | touched_a[b] | touched_v[b]
    //                            | a_neg[b*K] | v_neg[b*K] | acc[4]
    float* ws        = (float*)d_ws;
    float* cond_f    = ws;
    float* touched_a = cond_f + n;
    float* touched_v = touched_a + b;
    float* a_neg     = touched_v + b;
    float* v_neg     = a_neg + (size_t)b * KCNT;
    float* acc       = v_neg + (size_t)b * KCNT;

    const int zcount = 2 * b + 2 * b * KCNT + 4;     // everything after cond_f
    k_zero<<<512, 256, 0, stream>>>(touched_a, zcount);

    // one wave (32 lanes) per 16 pairs -> n/16 waves
    int waves  = n / 16;
    int blocks = (waves * 32 + 127) / 128;
    k_cond<<<blocks, 128, 0, stream>>>(target, aidx, vidx,
                                       cond_f, touched_a, touched_v, n);

    k_neg<<<n / TILE, THREADS, 0, stream>>>(all_prob, aidx, vidx, cond_f,
                                            a_neg, v_neg, b);

    k_pos_loss<<<b / TILE, THREADS, 0, stream>>>(all_prob, target,
                                                 touched_a, touched_v,
                                                 a_neg, v_neg, acc);

    k_fin<<<1, 1, 0, stream>>>(acc, (float*)d_out);
}
